// MultiHeadAttention_91070486544496
// MI455X (gfx1250) — matrix operand order repaired and checked by reading, not compiled, not run on hardware
//
#include <hip/hip_runtime.h>
#include <hip/hip_bf16.h>

typedef __attribute__((ext_vector_type(16))) _Float16 v16h;
typedef __attribute__((ext_vector_type(8)))  _Float16 v8h;
typedef __attribute__((ext_vector_type(8)))  float    v8f;
typedef __attribute__((ext_vector_type(4)))  float    v4f;
typedef __attribute__((ext_vector_type(4)))  unsigned int u32x4;
typedef __attribute__((ext_vector_type(8)))  int      i32x8;
typedef __attribute__((ext_vector_type(4)))  int      i32x4;

#define DEV static __device__ __forceinline__

// ---------------------------------------------------------------------------
// WMMA helpers (CDNA5: D = A(16x32,f16) * B(32x16,f16) + C(16x16,f32))
// ---------------------------------------------------------------------------
DEV v8f wmma16(v16h a, v16h b, v8f c) {
  return __builtin_amdgcn_wmma_f32_16x16x32_f16(
      /*neg_a=*/false, a, /*neg_b=*/false, b,
      /*c_mod=*/(short)0, c, /*reuse_a=*/false, /*reuse_b=*/false);
}

// A-fragment (16x32 f16). Row M = lane&15. Per ISA 7.12.2:
//   elems 0..7  -> k = (lane>>4)*8 + e ; elems 8..15 -> k = 16+(lane>>4)*8+(e-8)
DEV v16h loadA(const _Float16* rowPtr, int lane) {
  const int k0 = (lane >> 4) * 8;
  v8h lo = *(const v8h*)(rowPtr + k0);
  v8h hi = *(const v8h*)(rowPtr + k0 + 16);
  v16h r;
#pragma unroll
  for (int i = 0; i < 8; ++i) { r[i] = lo[i]; r[i + 8] = hi[i]; }
  return r;
}

// B-fragment (32x16 f16). Column N = lane&15, elems 0..7 -> K=(lane>>4)*8+e; elems 8..15 -> K=16+(lane>>4)*8+(e-8).
DEV v16h loadB(const _Float16* colPtr, int lane) {
  const int k0 = (lane >> 4) * 8;
  v8h lo = *(const v8h*)(colPtr + k0);
  v8h hi = *(const v8h*)(colPtr + k0 + 16);
  v16h r;
#pragma unroll
  for (int i = 0; i < 8; ++i) { r[i] = lo[i]; r[i + 8] = hi[i]; }
  return r;
}

// ---------------------------------------------------------------------------
// CDNA5 async global->LDS copy (ASYNCcnt-tracked). LDS dest VGPR holds the
// wave-relative LDS byte offset = addr[31:0] of the generic __shared__ ptr.
// ---------------------------------------------------------------------------
DEV void asyncLoad16(const void* ldsDst, const void* g) {
  asm volatile("global_load_async_to_lds_b128 %0, %1, off"
               :: "v"((unsigned)(unsigned long long)(uintptr_t)ldsDst),
                  "v"((unsigned long long)(uintptr_t)g)
               : "memory");
}
DEV void waitAsync() { asm volatile("s_wait_asynccnt 0x0" ::: "memory"); }

// ---------------------------------------------------------------------------
// Tensor Data Mover: 2-D tile load (rows x 64 halves) global->LDS.
// D# per ISA 8.3/8.4: group0 = {count=1 | lds_addr | global_addr | type=2},
// group1 = {data_size=2B, pad_enable, pad_interval=32dw (one 64-half row),
//           pad_amount=4dw (8 halves) -> LDS row stride 72 halves,
//           tensor_dim0=64, tensor_dim1=rows, tile 64x rows, dim0_stride=K}.
// Issued per-wave (EXEC ignored); tracked by TENSORcnt.
// ---------------------------------------------------------------------------
DEV void tdmLoadTile(const _Float16* gsrc, const void* ldsDst, int K, int rows) {
  const unsigned long long ga = (unsigned long long)(uintptr_t)gsrc;
  u32x4 g0;
  g0[0] = 1u;                                               // count=1
  g0[1] = (unsigned)(uintptr_t)ldsDst;                      // lds_addr (bytes)
  g0[2] = (unsigned)(ga & 0xFFFFFFFFu);                     // global_addr lo
  g0[3] = (unsigned)((ga >> 32) & 0x01FFFFFFu) | (2u << 30);// global hi | type=2
  i32x8 g1;
  g1[0] = (1 << 16) | (1 << 20) | (4 << 22) | (3 << 25);    // 2B | pad | 32dw | 4dw
  g1[1] = (int)(64u << 16);                                 // tensor_dim0 = 64
  g1[2] = (int)(((unsigned)rows & 0xFFFFu) << 16);          // tensor_dim1 = rows
  g1[3] = (int)(64u << 16);                                 // tile_dim0 = 64
  g1[4] = rows;                                             // tile_dim1 = rows
  g1[5] = K;                                                // dim0_stride lo
  g1[6] = 0;
  g1[7] = 0;
  i32x4 g2 = {0, 0, 0, 0};
  i32x4 g3 = {0, 0, 0, 0};
#if __clang_major__ >= 23
  i32x8 g4 = {0, 0, 0, 0, 0, 0, 0, 0};
  __builtin_amdgcn_tensor_load_to_lds(g0, g1, g2, g3, g4, 0);
#else
  __builtin_amdgcn_tensor_load_to_lds(g0, g1, g2, g3, 0);
#endif
}

// ---------------------------------------------------------------------------
// f32 -> f16 conversion (8 elements / thread, 16B vector ops)
// ---------------------------------------------------------------------------
__global__ __launch_bounds__(256) void cvt_f32_f16(const float* __restrict__ in,
                                                   _Float16* __restrict__ out,
                                                   int n8) {
  int i = blockIdx.x * blockDim.x + threadIdx.x;
  if (i >= n8) return;
  v4f a = ((const v4f*)in)[i * 2 + 0];
  v4f b = ((const v4f*)in)[i * 2 + 1];
  v8h r;
#pragma unroll
  for (int k = 0; k < 4; ++k) { r[k] = (_Float16)a[k]; r[k + 4] = (_Float16)b[k]; }
  ((v8h*)out)[i] = r;
}

// ---------------------------------------------------------------------------
// Y[M,N] = A[M,K](f16) * W[N,K](f16)^T. Block tile 128x128, BK=64.
// Double-buffered TDM staging: wave 0 issues one tensor_load per matrix per
// tile; everyone syncs on TENSORcnt + barrier; 16 WMMAs per K-tile per wave.
// ---------------------------------------------------------------------------
template <typename OutT>
__global__ __launch_bounds__(256) void gemm_xwT(const _Float16* __restrict__ A,
                                                const _Float16* __restrict__ W,
                                                OutT* __restrict__ Y,
                                                int M, int N, int K) {
  __shared__ __align__(16) _Float16 As[2][128][72];  // 64 halves + 8 pad (144B)
  __shared__ __align__(16) _Float16 Ws[2][128][72];

  const int tid   = threadIdx.x;
  const int lane  = tid & 31;
  const int wave  = tid >> 5;
  const int waveM = wave & 1;   // 0..1  -> 64 rows each
  const int waveN = wave >> 1;  // 0..3  -> 32 cols each
  const int m0 = blockIdx.x * 128;
  const int n0 = blockIdx.y * 128;

  auto issue = [&](int kb, int buf) {   // wave 0 only: 2 TDM descriptors
    tdmLoadTile(A + (size_t)m0 * K + kb, &As[buf][0][0], K, 128);
    tdmLoadTile(W + (size_t)n0 * K + kb, &Ws[buf][0][0], K, 128);
  };

  const v8f zacc = {0.f, 0.f, 0.f, 0.f, 0.f, 0.f, 0.f, 0.f};
  v8f acc[4][2];
#pragma unroll
  for (int mt = 0; mt < 4; ++mt)
#pragma unroll
    for (int nt = 0; nt < 2; ++nt) acc[mt][nt] = zacc;

  if (wave == 0) issue(0, 0);
  int cur = 0;
  for (int kb = 0; kb < K; kb += 64) {
    __builtin_amdgcn_s_wait_tensorcnt((short)0);  // wave0: tile arrived (others: 0)
    __syncthreads();                              // publish tile; prev compute done
    if (kb + 64 < K && wave == 0) issue(kb + 64, cur ^ 1);

#pragma unroll
    for (int s = 0; s < 2; ++s) {   // two 16x16x32 K-steps inside the tile
      v16h aF[4], bF[2];
#pragma unroll
      for (int mt = 0; mt < 4; ++mt)
        aF[mt] = loadA(&As[cur][waveM * 64 + mt * 16 + (lane & 15)][s * 32], lane);
#pragma unroll
      for (int nt = 0; nt < 2; ++nt)
        bF[nt] = loadB(&Ws[cur][waveN * 32 + nt * 16 + (lane & 15)][s * 32], lane);
#pragma unroll
      for (int mt = 0; mt < 4; ++mt)
#pragma unroll
        for (int nt = 0; nt < 2; ++nt)
          acc[mt][nt] = wmma16(aF[mt], bF[nt], acc[mt][nt]);
    }
    cur ^= 1;
  }

  // Epilogue: C layout -> M = j + 8*(lane>>4), N = lane&15
#pragma unroll
  for (int mt = 0; mt < 4; ++mt)
#pragma unroll
    for (int nt = 0; nt < 2; ++nt)
#pragma unroll
      for (int j = 0; j < 8; ++j) {
        int row = m0 + waveM * 64 + mt * 16 + j + ((lane >> 4) << 3);
        int col = n0 + waveN * 32 + nt * 16 + (lane & 15);
        Y[(size_t)row * N + col] = (OutT)acc[mt][nt][j];
      }
}

// ---------------------------------------------------------------------------
// Causal flash attention. Grid: (S/128, B*H). 8 waves; wave owns 16 q-rows.
// Q,K,V,O are f16 in [B,S,D] layout; head h uses columns h*64..h*64+63.
// K tile staged with async copies; V transposed into LDS via VGPR path.
// ---------------------------------------------------------------------------
__global__ __launch_bounds__(256) void attn_causal(const _Float16* __restrict__ Q,
                                                   const _Float16* __restrict__ Kx,
                                                   const _Float16* __restrict__ Vx,
                                                   _Float16* __restrict__ O) {
  constexpr int S = 2048, D = 1024, DK = 64;
  __shared__ __align__(16) _Float16 Ks[32][72];      // 32 keys x 64 dk (+pad)
  __shared__ __align__(16) _Float16 Vt[64][40];      // transposed: dk x keys
  __shared__ __align__(16) _Float16 Ps[8][16][40];   // per-wave P staging

  const int tid  = threadIdx.x;
  const int lane = tid & 31;
  const int wave = tid >> 5;
  const int qblk = blockIdx.x;      // 0..15
  const int bh   = blockIdx.y;      // 0..63
  const int b = bh >> 4, h = bh & 15;
  const int qBase = qblk * 128 + wave * 16;
  const size_t baseBS = (size_t)b * S;

  // Q fragments (dk 0..31 and 32..63), live in registers for the whole kernel
  const _Float16* qrow = Q + (baseBS + qBase + (lane & 15)) * D + h * DK;
  const v16h aQ0 = loadA(qrow, lane);
  const v16h aQ1 = loadA(qrow + 32, lane);

  const v8f zacc = {0.f, 0.f, 0.f, 0.f, 0.f, 0.f, 0.f, 0.f};
  float mrow[8], lrow[8];
  v8f o[4];
#pragma unroll
  for (int j = 0; j < 8; ++j) { mrow[j] = -1e30f; lrow[j] = 0.f; }
#pragma unroll
  for (int t = 0; t < 4; ++t) o[t] = zacc;

  const int keyEnd = qblk * 128 + 128;   // causal: block needs keys < keyEnd
  const float scale = 0.125f;            // 1/sqrt(64)
  const int halfSel = lane >> 4;

  for (int kb0 = 0; kb0 < keyEnd; kb0 += 32) {
    __syncthreads();                // previous tile's compute complete
    {  // stage K (async) and V (transposed, VGPR path)
      int key = tid >> 3;           // 0..31
      int seg = (tid & 7) * 8;      // 0..56 halves
      asyncLoad16(&Ks[key][seg], Kx + (baseBS + kb0 + key) * D + h * DK + seg);
      const _Float16* vg = Vx + (baseBS + kb0 + key) * D + h * DK + seg;
      v8h vv = *(const v8h*)vg;
#pragma unroll
      for (int i = 0; i < 8; ++i) Vt[seg + i][key] = vv[i];
    }
    waitAsync();
    __syncthreads();

    // S = Q * K^T : two 16x16 score tiles (keys 0..15 / 16..31), K-dim = 64
    v16h bK00 = loadB(&Ks[lane & 15][0],  lane);
    v16h bK01 = loadB(&Ks[lane & 15][32], lane);
    v16h bK10 = loadB(&Ks[16 + (lane & 15)][0],  lane);
    v16h bK11 = loadB(&Ks[16 + (lane & 15)][32], lane);
    v8f s0 = wmma16(aQ1, bK01, wmma16(aQ0, bK00, zacc));
    v8f s1 = wmma16(aQ1, bK11, wmma16(aQ0, bK10, zacc));

    const int colA = kb0 + (lane & 15);
    const int colB = kb0 + 16 + (lane & 15);
#pragma unroll
    for (int j = 0; j < 8; ++j) {
      const int qi = qBase + j + (halfSel << 3);
      float sa = s0[j] * scale;
      float sb = s1[j] * scale;
      if (colA > qi) sa = -1e30f;       // causal mask
      if (colB > qi) sb = -1e30f;
      float rmax = fmaxf(sa, sb);
#pragma unroll
      for (int d = 1; d < 16; d <<= 1) rmax = fmaxf(rmax, __shfl_xor(rmax, d, 32));
      const float mnew = fmaxf(mrow[j], rmax);
      const float corr = __expf(mrow[j] - mnew);
      const float pa = __expf(sa - mnew);
      const float pb = __expf(sb - mnew);
      float rsum = pa + pb;
#pragma unroll
      for (int d = 1; d < 16; d <<= 1) rsum += __shfl_xor(rsum, d, 32);
      lrow[j] = lrow[j] * corr + rsum;
      mrow[j] = mnew;
#pragma unroll
      for (int t = 0; t < 4; ++t) o[t][j] *= corr;
      // stage P (C layout -> row-major f16) for the A-fragment reload
      const int r = j + (halfSel << 3);
      Ps[wave][r][lane & 15]        = (_Float16)pa;
      Ps[wave][r][16 + (lane & 15)] = (_Float16)pb;
    }
    // wave-local LDS RAW (Ps is private to this wave): wait DS, no barrier
    asm volatile("s_wait_dscnt 0" ::: "memory");

    const v16h aP = loadA(&Ps[wave][lane & 15][0], lane);
#pragma unroll
    for (int t = 0; t < 4; ++t) {
      v16h bV = loadB(&Vt[t * 16 + (lane & 15)][0], lane);
      o[t] = wmma16(aP, bV, o[t]);
    }
  }

  // normalize and write ctx (f16, [B,S,D])
#pragma unroll
  for (int j = 0; j < 8; ++j) {
    const float rinv = 1.0f / lrow[j];
    const int qi = qBase + j + (halfSel << 3);
    _Float16* op = O + (baseBS + qi) * D + h * DK + (lane & 15);
#pragma unroll
    for (int t = 0; t < 4; ++t) op[t * 16] = (_Float16)(o[t][j] * rinv);
  }
}

// ---------------------------------------------------------------------------
// Launch: convert -> QKV GEMMs -> attention -> output GEMM
// ---------------------------------------------------------------------------
extern "C" void kernel_launch(void* const* d_in, const int* in_sizes, int n_in,
                              void* d_out, int out_size, void* d_ws, size_t ws_size,
                              hipStream_t stream) {
  (void)in_sizes; (void)n_in; (void)out_size; (void)ws_size;
  constexpr int Bc = 4, Sc = 2048, Dc = 1024;
  constexpr size_t nX = (size_t)Bc * Sc * Dc;   // 8,388,608
  constexpr size_t nW = (size_t)Dc * Dc;        // 1,048,576
  constexpr int M = Bc * Sc;                    // 8192

  const float* x  = (const float*)d_in[0];
  const float* WQ = (const float*)d_in[1];
  const float* WK = (const float*)d_in[2];
  const float* WV = (const float*)d_in[3];
  const float* WO = (const float*)d_in[4];
  float* out = (float*)d_out;

  _Float16* w   = (_Float16*)d_ws;   // ~88 MB of workspace used
  _Float16* hx  = w;
  _Float16* hwq = hx + nX;
  _Float16* hwk = hwq + nW;
  _Float16* hwv = hwk + nW;
  _Float16* hwo = hwv + nW;
  _Float16* q   = hwo + nW;
  _Float16* k   = q + nX;
  _Float16* v   = k + nX;
  _Float16* ctx = v + nX;

  cvt_f32_f16<<<dim3((unsigned)(nX / 8 / 256)), 256, 0, stream>>>(x,  hx,  (int)(nX / 8));
  cvt_f32_f16<<<dim3((unsigned)(nW / 8 / 256)), 256, 0, stream>>>(WQ, hwq, (int)(nW / 8));
  cvt_f32_f16<<<dim3((unsigned)(nW / 8 / 256)), 256, 0, stream>>>(WK, hwk, (int)(nW / 8));
  cvt_f32_f16<<<dim3((unsigned)(nW / 8 / 256)), 256, 0, stream>>>(WV, hwv, (int)(nW / 8));
  cvt_f32_f16<<<dim3((unsigned)(nW / 8 / 256)), 256, 0, stream>>>(WO, hwo, (int)(nW / 8));

  dim3 gGemm(M / 128, Dc / 128);  // 64 x 8
  gemm_xwT<_Float16><<<gGemm, 256, 0, stream>>>(hx, hwq, q, M, Dc, Dc);
  gemm_xwT<_Float16><<<gGemm, 256, 0, stream>>>(hx, hwk, k, M, Dc, Dc);
  gemm_xwT<_Float16><<<gGemm, 256, 0, stream>>>(hx, hwv, v, M, Dc, Dc);

  attn_causal<<<dim3(Sc / 128, Bc * 16), 256, 0, stream>>>(q, k, v, ctx);

  gemm_xwT<float><<<gGemm, 256, 0, stream>>>(ctx, hwo, out, M, Dc, Dc);
}
